// SE2ASGHead_33423435497892
// MI455X (gfx1250) — matrix operand order repaired and checked by reading, not compiled, not run on hardware
//
#include <hip/hip_runtime.h>
#include <hip/hip_bf16.h>
#include <math.h>

// ---------------------------------------------------------------------------
// Types
// ---------------------------------------------------------------------------
typedef __attribute__((ext_vector_type(16))) __bf16    v16bf;
typedef __attribute__((ext_vector_type(8)))  float     v8f;
typedef __attribute__((ext_vector_type(4))) unsigned   u32x4;
typedef __attribute__((ext_vector_type(4)))  float     f32x4;

union FragU { u32x4 u[2]; v16bf v; };

__device__ __forceinline__ unsigned short f2bf(float x) {
    union { float f; unsigned u; } v; v.f = x;
    unsigned r = v.u + 0x7FFFu + ((v.u >> 16) & 1u);   // round-to-nearest-even
    return (unsigned short)(r >> 16);
}
__device__ __forceinline__ float bf2f(unsigned short h) {
    union { float f; unsigned u; } v; v.u = ((unsigned)h) << 16; return v.f;
}

// A fragment (16x32 bf16, M x K) from a row-major LDS tile.
// CDNA5 ISA 7.12.2: per-lane elements 0..7 = K k0+hi*8+{0..7},
// elements 8..15 = K k0+16+hi*8+{0..7}; row M = lane%16.
__device__ __forceinline__ v16bf load_a_frag(const unsigned short* tile,
                                             int stride, int k0, int lane) {
    int m  = lane & 15;
    int hi = lane >> 4;
    const unsigned short* p = tile + m * stride + k0 + hi * 8;
    FragU f;
    f.u[0] = *(const u32x4*)(p);
    f.u[1] = *(const u32x4*)(p + 16);
    return f.v;
}

// B fragment (32x16 bf16, K x N) from packed weights Wt[n][k] (bf16, K-major).
// Lane = column n (lane%16); elements 0..7 = K k0+hi*8+{0..7}, elements 8..15 = K k0+16+hi*8+{0..7}.
__device__ __forceinline__ v16bf load_b_frag(const unsigned short* Wt,
                                             int Kdim, int ncol, int k0, int lane) {
    int n  = lane & 15;
    int hi = lane >> 4;
    const unsigned short* p = Wt + (size_t)(ncol + n) * Kdim + k0 + hi * 8;
    FragU f;
    f.u[0] = *(const u32x4*)(p);
    f.u[1] = *(const u32x4*)(p + 16);
    return f.v;
}

// ---------------------------------------------------------------------------
// Weight pack kernel: dst[n*K + k] = bf16(src[k*N + n])   (transpose + cvt)
// ---------------------------------------------------------------------------
__global__ void pack_wt_kernel(const float* __restrict__ src,
                               unsigned short* __restrict__ dst,
                               int K, int Nn) {
    int idx = blockIdx.x * blockDim.x + threadIdx.x;
    int total = K * Nn;
    if (idx >= total) return;
    int n = idx / K, k = idx - n * K;
    dst[idx] = f2bf(src[k * Nn + n]);
}

// ---------------------------------------------------------------------------
// Message-passing layer: ONE wave per block, TWO nodes per wave.
// Each B fragment is loaded once and consumed by two WMMAs (one per node),
// halving weight-fragment traffic per FLOP.   C=256, K=16, EH=128, HID=256.
// ---------------------------------------------------------------------------
__global__ __launch_bounds__(32) void msg_layer_kernel(
    const float* __restrict__ x_in, const float* __restrict__ boxes,
    const int* __restrict__ nbr_idx,
    const float* __restrict__ eW1, const float* __restrict__ eb1,
    const float* __restrict__ eb2, const float* __restrict__ mb1,
    const float* __restrict__ mb2, const float* __restrict__ aW,
    const float* __restrict__ ab,
    const unsigned short* __restrict__ eW2t,   // [128][128] bf16 (out-major)
    const unsigned short* __restrict__ mW1t,   // [256][384] bf16
    const unsigned short* __restrict__ mW2t,   // [256][256] bf16
    float* __restrict__ x_out, int N)
{
    // Per-node tiles; sM doubles as the h tile (disjoint live ranges).
    __shared__ __align__(16) unsigned short sNbr[2][16 * 256]; // aligned nbr feats
    __shared__ __align__(16) unsigned short sE  [2][16 * 128]; // e_emb
    __shared__ __align__(16) unsigned short sM  [2][16 * 256]; // h (16x128) then m
    __shared__ float sXi   [2][256];
    __shared__ float sEdge [2][16 * 4];
    __shared__ float sCS   [2][16 * 2];
    __shared__ float sAlpha[2][16];
    __shared__ int   sJ    [2][16];

    const int lane = threadIdx.x & 31;
    const int m16  = lane & 15;
    const int hi   = lane >> 4;

    int idx[2];
    idx[0] = blockIdx.x * 2;
    idx[1] = idx[0] + 1;
    if (idx[0] >= N) idx[0] = N - 1;           // duplicates write identical data
    if (idx[1] >= N) idx[1] = N - 1;

    // ---- stage 0: node features + per-neighbor edge features --------------
    for (int p = 0; p < 2; ++p) {
        int i = idx[p];
        for (int c = lane; c < 256; c += 32) sXi[p][c] = x_in[(size_t)i * 256 + c];
        if (lane < 16) {
            int k = lane;
            int j = nbr_idx[i * 16 + k];
            sJ[p][k] = j;
            float bix = boxes[i * 5 + 0], biy = boxes[i * 5 + 1];
            float biw = boxes[i * 5 + 2], bih = boxes[i * 5 + 3], bit = boxes[i * 5 + 4];
            float bjx = boxes[j * 5 + 0], bjy = boxes[j * 5 + 1];
            float bjw = boxes[j * 5 + 2], bjh = boxes[j * 5 + 3], bjt = boxes[j * 5 + 4];
            float dx = bix - bjx, dy = biy - bjy;
            float dist  = sqrtf(dx * dx + dy * dy);
            float ratio = fminf(biw, bih) / fminf(bjw, bjh);
            float scale = logf(fmaxf(ratio, 1e-6f));
            float dth   = bit - bjt;
            float cdt = cosf(dth), sdt = sinf(dth);
            sEdge[p][k * 4 + 0] = dist;
            sEdge[p][k * 4 + 1] = scale;
            sEdge[p][k * 4 + 2] = cdt;
            sEdge[p][k * 4 + 3] = sdt;
            sCS[p][k * 2 + 0] = cdt;
            sCS[p][k * 2 + 1] = sdt;
        }
    }
    __builtin_amdgcn_wave_barrier();

    // ---- stage 1: gather neighbors, SE(2)-align, write bf16 tiles ----------
    {
        int c0 = lane * 8;                                   // 8 channels/lane
        for (int p = 0; p < 2; ++p) {
            for (int k = 0; k < 16; ++k) {
                int j = sJ[p][k];
                const float* ptr = x_in + (size_t)j * 256 + c0;
                f32x4 a0 = *(const f32x4*)(ptr);
                f32x4 a1 = *(const f32x4*)(ptr + 4);
                if (lane == 0) {                             // channels 0,1 rotate
                    float c = sCS[p][k * 2 + 0], s = sCS[p][k * 2 + 1];
                    float x0 = a0.x, y0 = a0.y;
                    a0.x = c * x0 - s * y0;
                    a0.y = s * x0 + c * y0;
                }
                unsigned q0 = (unsigned)f2bf(a0.x) | ((unsigned)f2bf(a0.y) << 16);
                unsigned q1 = (unsigned)f2bf(a0.z) | ((unsigned)f2bf(a0.w) << 16);
                unsigned q2 = (unsigned)f2bf(a1.x) | ((unsigned)f2bf(a1.y) << 16);
                unsigned q3 = (unsigned)f2bf(a1.z) | ((unsigned)f2bf(a1.w) << 16);
                u32x4 pk = { q0, q1, q2, q3 };
                *(u32x4*)(&sNbr[p][k * 256 + c0]) = pk;
            }
        }
    }
    __builtin_amdgcn_wave_barrier();

    // ---- stage 2: h = relu(e @ eW1 + eb1) (K=4; VALU) -> sM[p][0 .. 16*128) -
    for (int p = 0; p < 2; ++p) {
        float e0 = sEdge[p][m16 * 4 + 0], e1 = sEdge[p][m16 * 4 + 1];
        float e2 = sEdge[p][m16 * 4 + 2], e3 = sEdge[p][m16 * 4 + 3];
        for (int t = 0; t < 64; ++t) {
            int c = hi * 64 + t;
            float acc = eb1[c]
                      + e0 * eW1[0 * 128 + c] + e1 * eW1[1 * 128 + c]
                      + e2 * eW1[2 * 128 + c] + e3 * eW1[3 * 128 + c];
            sM[p][m16 * 128 + c] = f2bf(fmaxf(acc, 0.f));
        }
    }
    __builtin_amdgcn_wave_barrier();

    // ---- stage 3: e_emb = relu(h @ eW2 + eb2)  (dual-node WMMA) ------------
    for (int nt = 0; nt < 8; ++nt) {
        v8f acc0 = {}, acc1 = {};
#pragma unroll
        for (int kt = 0; kt < 4; ++kt) {
            v16bf b  = load_b_frag(eW2t, 128, nt * 16, kt * 32, lane);
            v16bf a0 = load_a_frag(sM[0], 128, kt * 32, lane);
            v16bf a1 = load_a_frag(sM[1], 128, kt * 32, lane);
            acc0 = __builtin_amdgcn_wmma_f32_16x16x32_bf16(
                false, a0, false, b, (short)0, acc0, false, false);
            acc1 = __builtin_amdgcn_wmma_f32_16x16x32_bf16(
                false, a1, false, b, (short)0, acc1, false, false);
        }
        int col = nt * 16 + m16;
        float bias = eb2[col];
#pragma unroll
        for (int v = 0; v < 8; ++v) {
            int M = v + hi * 8;
            sE[0][M * 128 + col] = f2bf(fmaxf(acc0[v] + bias, 0.f));
            sE[1][M * 128 + col] = f2bf(fmaxf(acc1[v] + bias, 0.f));
        }
    }
    __builtin_amdgcn_wave_barrier();

    // ---- stage 4: attention logits + softmax over K=16 ---------------------
    for (int p = 0; p < 2; ++p) {
        float acc = 0.f;
        if (hi == 0) {  // idx 0..319: q[0..255] + nbr[0..63]
            for (int c = 0; c < 256; ++c) acc += sXi[p][c] * aW[c];
            for (int c = 0; c < 64;  ++c) acc += bf2f(sNbr[p][m16 * 256 + c]) * aW[256 + c];
        } else {        // idx 320..639: nbr[64..255] + e_emb[0..127]
            for (int c = 64; c < 256; ++c) acc += bf2f(sNbr[p][m16 * 256 + c]) * aW[256 + c];
            for (int c = 0;  c < 128; ++c) acc += bf2f(sE[p][m16 * 128 + c])   * aW[512 + c];
        }
        acc += __shfl_xor(acc, 16, 32);
        float logit = acc + ab[0];
        float mx = logit;
#pragma unroll
        for (int off = 1; off < 16; off <<= 1) mx = fmaxf(mx, __shfl_xor(mx, off, 32));
        float ex = expf(logit - mx);
        float sm = ex;
#pragma unroll
        for (int off = 1; off < 16; off <<= 1) sm += __shfl_xor(sm, off, 32);
        if (lane < 16) sAlpha[p][m16] = ex / sm;
    }
    __builtin_amdgcn_wave_barrier();

    // ---- stage 5: m = relu([nbr | e_emb] @ mW1 + mb1)  (dual-node WMMA) ----
    for (int nt = 0; nt < 16; ++nt) {
        v8f acc0 = {}, acc1 = {};
#pragma unroll
        for (int kt = 0; kt < 12; ++kt) {
            v16bf b = load_b_frag(mW1t, 384, nt * 16, kt * 32, lane);
            v16bf a0, a1;
            if (kt < 8) {
                a0 = load_a_frag(sNbr[0], 256, kt * 32, lane);
                a1 = load_a_frag(sNbr[1], 256, kt * 32, lane);
            } else {
                a0 = load_a_frag(sE[0], 128, (kt - 8) * 32, lane);
                a1 = load_a_frag(sE[1], 128, (kt - 8) * 32, lane);
            }
            acc0 = __builtin_amdgcn_wmma_f32_16x16x32_bf16(
                false, a0, false, b, (short)0, acc0, false, false);
            acc1 = __builtin_amdgcn_wmma_f32_16x16x32_bf16(
                false, a1, false, b, (short)0, acc1, false, false);
        }
        int col = nt * 16 + m16;
        float bias = mb1[col];
#pragma unroll
        for (int v = 0; v < 8; ++v) {
            int M = v + hi * 8;
            sM[0][M * 256 + col] = f2bf(fmaxf(acc0[v] + bias, 0.f));
            sM[1][M * 256 + col] = f2bf(fmaxf(acc1[v] + bias, 0.f));
        }
    }
    __builtin_amdgcn_wave_barrier();

    // ---- stage 6: msg = m @ mW2 + mb2 ; out = x + sum_k alpha_k * msg_k ----
    {
        float al0[8], al1[8];
#pragma unroll
        for (int v = 0; v < 8; ++v) {
            al0[v] = sAlpha[0][v + hi * 8];
            al1[v] = sAlpha[1][v + hi * 8];
        }
        for (int nt = 0; nt < 16; ++nt) {
            v8f acc0 = {}, acc1 = {};
#pragma unroll
            for (int kt = 0; kt < 8; ++kt) {
                v16bf b  = load_b_frag(mW2t, 256, nt * 16, kt * 32, lane);
                v16bf a0 = load_a_frag(sM[0], 256, kt * 32, lane);
                v16bf a1 = load_a_frag(sM[1], 256, kt * 32, lane);
                acc0 = __builtin_amdgcn_wmma_f32_16x16x32_bf16(
                    false, a0, false, b, (short)0, acc0, false, false);
                acc1 = __builtin_amdgcn_wmma_f32_16x16x32_bf16(
                    false, a1, false, b, (short)0, acc1, false, false);
            }
            int col = nt * 16 + m16;
            float bias = mb2[col];
            float p0 = 0.f, p1 = 0.f;
#pragma unroll
            for (int v = 0; v < 8; ++v) {
                p0 += al0[v] * (acc0[v] + bias);
                p1 += al1[v] * (acc1[v] + bias);
            }
            p0 += __shfl_xor(p0, 16, 32);
            p1 += __shfl_xor(p1, 16, 32);
            if (hi == 0) {
                x_out[(size_t)idx[0] * 256 + col] = sXi[0][col] + p0;
                x_out[(size_t)idx[1] * 256 + col] = sXi[1][col] + p1;
            }
        }
    }
}

// ---------------------------------------------------------------------------
// Head: delta/apply_deltas + score MLP + copy x.  One block (128 thr) / node.
// ---------------------------------------------------------------------------
__global__ __launch_bounds__(128) void head_kernel(
    const float* __restrict__ x, const float* __restrict__ boxes,
    const float* __restrict__ dW, const float* __restrict__ db,
    const float* __restrict__ sW1, const float* __restrict__ sb1,
    const float* __restrict__ sW2, const float* __restrict__ sb2,
    float* __restrict__ out_boxes, float* __restrict__ out_score,
    float* __restrict__ out_x, int N)
{
    __shared__ float sX[256];
    __shared__ float sS[128];
    __shared__ float sD[6];
    int i = blockIdx.x;
    if (i >= N) return;
    int t = threadIdx.x;

    sX[t]       = x[(size_t)i * 256 + t];
    sX[t + 128] = x[(size_t)i * 256 + t + 128];
    __syncthreads();

    float acc = sb1[t];
    for (int c = 0; c < 256; ++c) acc += sX[c] * sW1[c * 128 + t];
    sS[t] = fmaxf(acc, 0.f) * sW2[t];

    out_x[(size_t)i * 256 + t]       = sX[t];
    out_x[(size_t)i * 256 + t + 128] = sX[t + 128];

    if (t < 6) {
        float a = db[t];
        for (int c = 0; c < 256; ++c) a += sX[c] * dW[c * 6 + t];
        sD[t] = a;
    }
    __syncthreads();
    for (int off = 64; off > 0; off >>= 1) {
        if (t < off) sS[t] += sS[t + off];
        __syncthreads();
    }
    if (t == 0) {
        out_score[i] = sS[0] + sb2[0];
        float bx = boxes[i * 5 + 0], by = boxes[i * 5 + 1];
        float bw = boxes[i * 5 + 2], bh = boxes[i * 5 + 3], th = boxes[i * 5 + 4];
        float w_ = fmaxf(bw * (1.f + tanhf(sD[2])), 0.001f);
        float h_ = fmaxf(bh * (1.f + tanhf(sD[3])), 0.001f);
        float vx = cosf(th) + sD[4];
        float vy = sinf(th) + sD[5];
        out_boxes[i * 5 + 0] = bx + sD[0];
        out_boxes[i * 5 + 1] = by + sD[1];
        out_boxes[i * 5 + 2] = w_;
        out_boxes[i * 5 + 3] = h_;
        out_boxes[i * 5 + 4] = atan2f(vy, vx);
    }
}

// ---------------------------------------------------------------------------
// Launch
// ---------------------------------------------------------------------------
extern "C" void kernel_launch(void* const* d_in, const int* in_sizes, int n_in,
                              void* d_out, int out_size, void* d_ws, size_t ws_size,
                              hipStream_t stream)
{
    const float* roi   = (const float*)d_in[0];
    const float* boxes = (const float*)d_in[1];
    const int*   nbr   = (const int*)  d_in[3];
    const int N = in_sizes[2];                 // scores has N elements

    // params flattening: layers[0..2] x {eW1,eb1,eW2,eb2,mW1,mb1,mW2,mb2,aW,ab}
    // then dW, db, sW1, sb1, sW2, sb2
    const int PB = 4;                          // params base index
    const float* dW  = (const float*)d_in[PB + 30 + 0];
    const float* db  = (const float*)d_in[PB + 30 + 1];
    const float* sW1 = (const float*)d_in[PB + 30 + 2];
    const float* sb1 = (const float*)d_in[PB + 30 + 3];
    const float* sW2 = (const float*)d_in[PB + 30 + 4];
    const float* sb2 = (const float*)d_in[PB + 30 + 5];

    // workspace: x ping/pong (f32) then packed bf16 weights per layer
    float* xA = (float*)d_ws;
    float* xB = xA + (size_t)N * 256;
    unsigned short* wpack = (unsigned short*)(xB + (size_t)N * 256);
    const int P_EW2 = 128 * 128, P_MW1 = 384 * 256, P_MW2 = 256 * 256;
    const int P_LAYER = P_EW2 + P_MW1 + P_MW2;

    // pack weights (transpose to [out][in], cvt to bf16)
    for (int l = 0; l < 3; ++l) {
        const float* eW2 = (const float*)d_in[PB + 10 * l + 2];
        const float* mW1 = (const float*)d_in[PB + 10 * l + 4];
        const float* mW2 = (const float*)d_in[PB + 10 * l + 6];
        unsigned short* wl = wpack + (size_t)l * P_LAYER;
        pack_wt_kernel<<<(P_EW2 + 255) / 256, 256, 0, stream>>>(eW2, wl, 128, 128);
        pack_wt_kernel<<<(P_MW1 + 255) / 256, 256, 0, stream>>>(mW1, wl + P_EW2, 384, 256);
        pack_wt_kernel<<<(P_MW2 + 255) / 256, 256, 0, stream>>>(mW2, wl + P_EW2 + P_MW1, 256, 256);
    }

    float* out        = (float*)d_out;
    float* out_boxes  = out;
    float* out_score  = out + (size_t)N * 5;
    float* out_x      = out + (size_t)N * 6;

    const float* xin = roi;
    float* xouts[3] = { xA, xB, out_x };       // final layer writes x into d_out
    const int blocks = (N + 1) / 2;            // one wave / block, 2 nodes / wave
    for (int l = 0; l < 3; ++l) {
        const int b = PB + 10 * l;
        unsigned short* wl = wpack + (size_t)l * P_LAYER;
        msg_layer_kernel<<<blocks, 32, 0, stream>>>(
            xin, boxes, nbr,
            (const float*)d_in[b + 0],  // eW1
            (const float*)d_in[b + 1],  // eb1
            (const float*)d_in[b + 3],  // eb2
            (const float*)d_in[b + 5],  // mb1
            (const float*)d_in[b + 7],  // mb2
            (const float*)d_in[b + 8],  // aW
            (const float*)d_in[b + 9],  // ab
            wl, wl + P_EW2, wl + P_EW2 + P_MW1,
            xouts[l], N);
        xin = xouts[l];
    }

    head_kernel<<<N, 128, 0, stream>>>(out_x, boxes, dW, db, sW1, sb1, sW2, sb2,
                                       out_boxes, out_score, out_x, N);
}
